// KerasLMU_46806553591872
// MI455X (gfx1250) — matrix operand order repaired and checked by reading, not compiled, not run on hardware
//
#include <hip/hip_runtime.h>

// ---------------- problem constants ----------------
#define ORDER   256
#define HIDDEN  512
#define FEAT    128
#define BATCH   64
#define SEQ     1025
#define STEPS   1024
#define SLOPE   0.2f

#define NWG      8          // grid workgroups; each owns 64 hidden columns
#define NTHREADS 512        // 16 waves (wave32)

typedef _Float16 v8h  __attribute__((ext_vector_type(8)));
typedef _Float16 v16h __attribute__((ext_vector_type(16)));
typedef float    v8f  __attribute__((ext_vector_type(8)));
typedef float    v4f  __attribute__((ext_vector_type(4)));

// ---------------- workspace layout (bytes) ----------------
#define WS_CTR  0
#define WS_H0   256
#define WS_H1   (WS_H0 + BATCH*HIDDEN*2)
#define WS_A    (WS_H1 + BATCH*HIDDEN*2)          // swizzled A^T   (K=256,N=256)
#define WS_WX   (WS_A  + ORDER*ORDER*2)           // swizzled Wx    (K=128,N=512)
#define WS_WH   (WS_WX + FEAT*HIDDEN*2)           // swizzled Wh^T  (K=512,N=512)
#define WS_WM   (WS_WH + HIDDEN*HIDDEN*2)         // swizzled Wm^T  (K=256,N=512)

// B-fragment layout of a 32x16 f16 tile (ISA 7.12.2):
//   lanes 0-15 hold K=0..7,16..23 of column N=lane; lanes 16-31 hold K=8..15,24..31 of
//   column N=lane-16. Stored so each lane's 16 halves are 32 contiguous bytes.
__device__ __forceinline__ size_t bfrag_index(int k, int n, int nK) {
    int kt = k >> 5, nt = n >> 4;
    int kin = k & 31, nin = n & 15;
    int lane = ((kin >> 3) & 1) ? (nin + 16) : nin;
    int j = (kin & 7) + ((kin >> 4) << 3);
    return (((size_t)(nt * nK + kt)) << 9) + (size_t)lane * 16 + j; // halves
}

__global__ void k_swizzleB(const float* __restrict__ src, _Float16* __restrict__ dst,
                           int K, int N, int ldsrc, int trans) {
    int total = K * N;
    for (int idx = blockIdx.x * blockDim.x + threadIdx.x; idx < total;
         idx += gridDim.x * blockDim.x) {
        int k = idx / N, n = idx - k * N;
        float v = trans ? src[(size_t)n * ldsrc + k] : src[(size_t)k * ldsrc + n];
        dst[bfrag_index(k, n, K >> 5)] = (_Float16)v;
    }
}

__global__ void k_init(unsigned* ctr, _Float16* h0, _Float16* h1) {
    int idx = blockIdx.x * blockDim.x + threadIdx.x;
    if (idx < 64) ctr[idx] = 0u;
    for (int i = idx; i < BATCH * HIDDEN; i += gridDim.x * blockDim.x) {
        h0[i] = (_Float16)0.0f;
        h1[i] = (_Float16)0.0f;
    }
}

// ---------------- fragment loaders ----------------
// A-matrix 16x32 f16 layout (ISA 7.12.2): lane<16 row=lane, halves K=k0+0..7 and
// k0+16..23; lane>=16 row=lane-16, halves K=k0+8..15 and k0+24..31.
__device__ __forceinline__ v16h load_a_f16(const _Float16* base /*[row0][k0]*/,
                                           int ld, int lane) {
    int r  = lane & 15;
    int kb = (lane >> 4) << 3;            // 0 or 8
    const _Float16* p = base + (size_t)r * ld + kb;
    v8h lo = *(const v8h*)p;
    v8h hi = *(const v8h*)(p + 16);
    v16h a;
#pragma unroll
    for (int i = 0; i < 8; ++i) { a[i] = lo[i]; a[8 + i] = hi[i]; }
    return a;
}

// same layout, source is f32 row-major (x), converted in VALU
__device__ __forceinline__ v16h load_a_f32(const float* base /*[row0][k0]*/,
                                           long ld, int lane) {
    int r  = lane & 15;
    int kb = (lane >> 4) << 3;
    const float* p = base + (long)r * ld + kb;
    v4f f0 = *(const v4f*)p;
    v4f f1 = *(const v4f*)(p + 4);
    v4f f2 = *(const v4f*)(p + 16);
    v4f f3 = *(const v4f*)(p + 20);
    v16h a;
#pragma unroll
    for (int i = 0; i < 4; ++i) {
        a[i]      = (_Float16)f0[i];
        a[4 + i]  = (_Float16)f1[i];
        a[8 + i]  = (_Float16)f2[i];
        a[12 + i] = (_Float16)f3[i];
    }
    return a;
}

__device__ __forceinline__ v16h load_b(const _Float16* frag, int lane) {
    const _Float16* p = frag + lane * 16;
    v8h lo = *(const v8h*)p;
    v8h hi = *(const v8h*)(p + 8);
    v16h b;
#pragma unroll
    for (int i = 0; i < 8; ++i) { b[i] = lo[i]; b[8 + i] = hi[i]; }
    return b;
}

__device__ __forceinline__ v8f wmma16(v16h a, v16h b, v8f c) {
    return __builtin_amdgcn_wmma_f32_16x16x32_f16(false, a, false, b,
                                                  (short)0, c, false, false);
}

// unified h-phase fragment selectors (kk = 0..27); statically resolved on unroll
__device__ __forceinline__ v16h h_a(int kk, const float* xb, long xld,
                                    const _Float16* hb, const _Float16* mb,
                                    int lane) {
    if (kk < 4)  return load_a_f32(xb + kk * 32, xld, lane);
    if (kk < 20) return load_a_f16(hb + (kk - 4) * 32, HIDDEN, lane);
    return load_a_f16(mb + (kk - 20) * 32, ORDER, lane);
}
__device__ __forceinline__ v16h h_b(int kk, const _Float16* wx, const _Float16* wh,
                                    const _Float16* wm, int gnt, int lane) {
    if (kk < 4)  return load_b(wx + (((size_t)(gnt * 4  + kk))        << 9), lane);
    if (kk < 20) return load_b(wh + (((size_t)(gnt * 16 + (kk - 4)))  << 9), lane);
    return load_b(wm + (((size_t)(gnt * 8 + (kk - 20))) << 9), lane);
}

// ---------------- main recurrent kernel ----------------
__global__ __launch_bounds__(NTHREADS, 1)
void k_lmu(const float* __restrict__ x, const float* __restrict__ Bv,
           const float* __restrict__ e_x,
           const _Float16* __restrict__ A_sw,  const _Float16* __restrict__ Wx_sw,
           const _Float16* __restrict__ Wh_sw, const _Float16* __restrict__ Wm_sw,
           _Float16* h0buf, _Float16* h1buf, unsigned* ctr,
           float* __restrict__ out) {
    __shared__ _Float16 m_lds[BATCH][ORDER];   // 32 KB private LMU state copy
    __shared__ float    u_part[BATCH][8];      // 2 KB partial dot products
    __shared__ float    u_lds[BATCH];
    __shared__ float    ex_lds[FEAT];

    const int tid  = threadIdx.x;
    const int lane = tid & 31;
    const int w    = tid >> 5;
    const int ng   = blockIdx.x;

    // zero m state; cache e_x in LDS
    for (int i = tid; i < BATCH * ORDER; i += NTHREADS) (&m_lds[0][0])[i] = (_Float16)0.0f;
    if (tid < FEAT) ex_lds[tid] = e_x[tid];
    __syncthreads();

    const _Float16* hprev = h0buf;
    _Float16*       hnext = h1buf;

    for (int t = 0; t < STEPS; ++t) {
        // ---- u = x_t @ e_x : 8 threads per batch row, fixed-order reduce ----
        {
            int b = tid >> 3, j = tid & 7;
            const float* xr = x + ((size_t)b * SEQ + (t + 1)) * FEAT + j * 16;
            float s = 0.0f;
#pragma unroll
            for (int f = 0; f < 16; ++f) s += xr[f] * ex_lds[j * 16 + f];
            u_part[b][j] = s;
            if (j == 0) __builtin_prefetch(xr + FEAT, 0, 0);   // next step's x row
        }
        __syncthreads();
        if (tid < BATCH) {
            float s = 0.0f;
#pragma unroll
            for (int j = 0; j < 8; ++j) s += u_part[tid][j];
            u_lds[tid] = s;
        }
        __syncthreads();

        // ---- phase 1: m_new = m_old @ A^T + b ⊗ u ----
        // wave w owns N-tile Nt = w (16 cols), all 4 M-tiles; kt-outer keeps
        // only ~2 A-frags + 2 B-frags live (B streams from L2-resident A_sw).
        v8f cm[4] = {v8f{}, v8f{}, v8f{}, v8f{}};
        {
            const _Float16* Ab = A_sw + (((size_t)(w * (ORDER / 32))) << 9);
            v16h bC = load_b(Ab, lane);
            v16h aC = load_a_f16(&m_lds[0][0], ORDER, lane);
#pragma unroll
            for (int kt = 0; kt < ORDER / 32; ++kt) {
                v16h bN = bC;
                if (kt < ORDER / 32 - 1)
                    bN = load_b(Ab + (((size_t)(kt + 1)) << 9), lane);
#pragma unroll
                for (int Mt = 0; Mt < 4; ++Mt) {
                    v16h aN = aC;
                    if (Mt < 3)
                        aN = load_a_f16(&m_lds[(Mt + 1) * 16][kt * 32], ORDER, lane);
                    else if (kt < ORDER / 32 - 1)
                        aN = load_a_f16(&m_lds[0][(kt + 1) * 32], ORDER, lane);
                    cm[Mt] = wmma16(aC, bC, cm[Mt]);
                    aC = aN;
                }
                bC = bN;
            }
        }
        const int mcol = w * 16 + (lane & 15);
        const float br = Bv[mcol];
#pragma unroll
        for (int Mt = 0; Mt < 4; ++Mt) {
            int row0 = Mt * 16 + ((lane >> 4) << 3);
#pragma unroll
            for (int r = 0; r < 8; ++r) cm[Mt][r] += br * u_lds[row0 + r];
        }
        __syncthreads();                          // all reads of m_old done
#pragma unroll
        for (int Mt = 0; Mt < 4; ++Mt) {
            int row0 = Mt * 16 + ((lane >> 4) << 3);
#pragma unroll
            for (int r = 0; r < 8; ++r)
                m_lds[row0 + r][mcol] = (_Float16)cm[Mt][r];
        }
        __syncthreads();                          // m_new visible

        // ---- barrier WAIT (deferred): h_{t-1} from all WGs must be visible.
        // The u-compute and whole m-recurrence above ran under barrier latency.
        if (t > 0) {
            if (tid == 0) {
                unsigned target = (unsigned)t * NWG;
                while (atomicAdd(ctr, 0u) < target) __builtin_amdgcn_s_sleep(2);
            }
            __syncthreads();
            __threadfence();
        }

        // ---- phase 2: h tile, unified 28-step pipelined K-loop ----
        {
            int Mt = w & 3, Nt = w >> 2;
            int gnt = ng * 4 + Nt;                // global 16-wide n-tile
            const float*    xb = x + ((size_t)(Mt * 16) * SEQ + (t + 1)) * FEAT;
            const _Float16* hb = hprev + (size_t)(Mt * 16) * HIDDEN;
            const _Float16* mb = &m_lds[Mt * 16][0];
            const long      xld = (long)SEQ * FEAT;

            v8f c = {};
            v16h aC = h_a(0, xb, xld, hb, mb, lane);
            v16h bC = h_b(0, Wx_sw, Wh_sw, Wm_sw, gnt, lane);
#pragma unroll
            for (int kk = 0; kk < 28; ++kk) {
                v16h aN = aC, bN = bC;
                if (kk < 27) {
                    aN = h_a(kk + 1, xb, xld, hb, mb, lane);
                    bN = h_b(kk + 1, Wx_sw, Wh_sw, Wm_sw, gnt, lane);
                }
                c = wmma16(aC, bC, c);
                aC = aN; bC = bN;
            }

            int row0 = Mt * 16 + ((lane >> 4) << 3);
            int col  = gnt * 16 + (lane & 15);
#pragma unroll
            for (int r = 0; r < 8; ++r) {
                float v = c[r];
                v = (v >= 0.0f) ? v : SLOPE * v;   // leaky relu
                out[((size_t)(row0 + r) * STEPS + t) * HIDDEN + col] = v;
                hnext[(size_t)(row0 + r) * HIDDEN + col] = (_Float16)v;
            }
        }

        // ---- barrier ARRIVE (release h_t) ----
        __threadfence();
        __syncthreads();
        if (tid == 0) atomicAdd(ctr, 1u);

        const _Float16* tp = hprev; hprev = hnext; hnext = (_Float16*)tp;
    }
}

// ---------------- launch ----------------
extern "C" void kernel_launch(void* const* d_in, const int* in_sizes, int n_in,
                              void* d_out, int out_size, void* d_ws, size_t ws_size,
                              hipStream_t stream) {
    const float* x  = (const float*)d_in[0];
    const float* A  = (const float*)d_in[1];
    const float* Bv = (const float*)d_in[2];
    const float* Wx = (const float*)d_in[3];
    const float* ex = (const float*)d_in[4];
    const float* Wh = (const float*)d_in[5];
    const float* Wm = (const float*)d_in[6];

    char* ws = (char*)d_ws;
    unsigned*  ctr   = (unsigned*)(ws + WS_CTR);
    _Float16*  h0    = (_Float16*)(ws + WS_H0);
    _Float16*  h1    = (_Float16*)(ws + WS_H1);
    _Float16*  A_sw  = (_Float16*)(ws + WS_A);
    _Float16*  Wx_sw = (_Float16*)(ws + WS_WX);
    _Float16*  Wh_sw = (_Float16*)(ws + WS_WH);
    _Float16*  Wm_sw = (_Float16*)(ws + WS_WM);

    k_init<<<64, 256, 0, stream>>>(ctr, h0, h1);
    // B[k][n] = A[n][k] ; W_h[n][k] ; W_m[n][k] ; W_x[k][n]
    k_swizzleB<<<256, 256, 0, stream>>>(A,  A_sw,  ORDER,  ORDER,  ORDER,  1);
    k_swizzleB<<<256, 256, 0, stream>>>(Wx, Wx_sw, FEAT,   HIDDEN, HIDDEN, 0);
    k_swizzleB<<<512, 256, 0, stream>>>(Wh, Wh_sw, HIDDEN, HIDDEN, HIDDEN, 1);
    k_swizzleB<<<256, 256, 0, stream>>>(Wm, Wm_sw, ORDER,  HIDDEN, ORDER,  1);

    k_lmu<<<NWG, NTHREADS, 0, stream>>>(x, Bv, ex, A_sw, Wx_sw, Wh_sw, Wm_sw,
                                        h0, h1, ctr, (float*)d_out);
}
